// MHAttention_61890478736071
// MI455X (gfx1250) — matrix operand order repaired and checked by reading, not compiled, not run on hardware
//
#include <hip/hip_runtime.h>
#include <hip/hip_bf16.h>
#include <cstdint>

// ---------------------------------------------------------------------------
// MHA forward for MI455X (gfx1250, wave32, WMMA).
// B=2, S=2048, D=1024, H=16, HD=64.
// GEMMs: v_wmma_f32_16x16x32_bf16, 32x32 register-blocked per wave.
// Attention: block-shared K/V staging via global_load_async_to_lds_b128.
// ---------------------------------------------------------------------------

#define BB 2
#define SS 2048
#define DD 1024
#define HH 16
#define HD 64

typedef __bf16 bf16;
typedef __attribute__((ext_vector_type(8)))  __bf16 v8bf;
typedef __attribute__((ext_vector_type(16))) __bf16 v16bf;
typedef __attribute__((ext_vector_type(8)))  float  v8f;

__device__ __forceinline__ v16bf cat16(v8bf lo, v8bf hi) {
  return __builtin_shufflevector(lo, hi, 0,1,2,3,4,5,6,7,8,9,10,11,12,13,14,15);
}

// A-matrix fragment (16x32 bf16, MxK). Lane l%16 = row M; g = l/16 selects the
// interleaved K chunks: elems 0-7 <- K[8g..8g+7], elems 8-15 <- K[16+8g..23+8g].
__device__ __forceinline__ v16bf load_a_frag(const bf16* row, int g) {
  v8bf lo = *(const v8bf*)(row + 8 * g);
  v8bf hi = *(const v8bf*)(row + 16 + 8 * g);
  return cat16(lo, hi);
}

// B-matrix fragment (32x16 bf16, KxN). Lane l%16 = col N; g = l/16 selects the
// interleaved K chunks: elems 0-7 <- K[8g..8g+7], elems 8-15 <- K[16+8g..23+8g].
__device__ __forceinline__ v16bf load_b_frag(const bf16* p, int g) {
  v8bf lo = *(const v8bf*)(p + 8 * g);
  v8bf hi = *(const v8bf*)(p + 16 + 8 * g);
  return cat16(lo, hi);
}

__device__ __forceinline__ v8f wmma_bf16(v16bf a, v16bf b, v8f c) {
  return __builtin_amdgcn_wmma_f32_16x16x32_bf16(false, a, false, b,
                                                 (short)0, c, false, false);
}

// CDNA5 async copy: LDS[lds_off] <- MEM[gp], 16 bytes, tracked by ASYNCcnt.
__device__ __forceinline__ void async_copy16(uint32_t lds_off, const bf16* gp) {
  asm volatile("global_load_async_to_lds_b128 %0, %1, off"
               :: "v"(lds_off), "v"(gp) : "memory");
}
__device__ __forceinline__ void wait_async0() {
  asm volatile("s_wait_asynccnt 0x0" ::: "memory");
}

// ---------------------------------------------------------------------------
// fp32 -> bf16 conversion
// ---------------------------------------------------------------------------
__global__ void cvt_f32_bf16(const float* __restrict__ src,
                             bf16* __restrict__ dst, int n) {
  int i = blockIdx.x * blockDim.x + threadIdx.x;
  if (i < n) dst[i] = (bf16)src[i];
}

// ---------------------------------------------------------------------------
// Y = X @ W.T   X:[M,K] bf16, W:[N,K] bf16. One wave per 32x32 tile
// (4 accumulators; A/B fragments reused across two WMMA each).
// ---------------------------------------------------------------------------
__global__ void gemm_xwt(const bf16* __restrict__ X, const bf16* __restrict__ W,
                         bf16* __restrict__ Yb, float* __restrict__ Yf,
                         int M, int N, int K) {
  int wave = (blockIdx.x * blockDim.x + threadIdx.x) >> 5;
  int lane = threadIdx.x & 31;
  int g = lane >> 4, l = lane & 15;
  int tilesN = N >> 5;
  int tm = wave / tilesN, tn = wave % tilesN;
  if (tm >= (M >> 5)) return;  // wave-uniform guard

  const bf16* a0 = X + (size_t)(tm * 32 + l) * K;
  const bf16* a1 = a0 + (size_t)16 * K;
  const bf16* b0 = W + (size_t)(tn * 32 + l) * K;
  const bf16* b1 = b0 + (size_t)16 * K;

  v8f c00 = {}, c01 = {}, c10 = {}, c11 = {};
#pragma unroll 2
  for (int kb = 0; kb < K; kb += 32) {
    __builtin_prefetch(a0 + kb + 128, 0, 1);  // -> global_prefetch_b8
    __builtin_prefetch(b0 + kb + 128, 0, 1);
    v16bf A0 = load_a_frag(a0 + kb, g);
    v16bf A1 = load_a_frag(a1 + kb, g);
    v16bf B0 = load_b_frag(b0 + kb, g);
    v16bf B1 = load_b_frag(b1 + kb, g);
    c00 = wmma_bf16(A0, B0, c00);
    c01 = wmma_bf16(A0, B1, c01);
    c10 = wmma_bf16(A1, B0, c10);
    c11 = wmma_bf16(A1, B1, c11);
  }
  int r0 = tm * 32 + 8 * g;  // C/D layout: VGPR r -> row r + 8g
  int n0 = tn * 32 + l, n1 = n0 + 16;
  if (Yf) {
#pragma unroll
    for (int r = 0; r < 8; ++r) {
      Yf[(size_t)(r0 + r) * N + n0]      = c00[r];
      Yf[(size_t)(r0 + r) * N + n1]      = c01[r];
      Yf[(size_t)(r0 + 16 + r) * N + n0] = c10[r];
      Yf[(size_t)(r0 + 16 + r) * N + n1] = c11[r];
    }
  } else {
#pragma unroll
    for (int r = 0; r < 8; ++r) {
      Yb[(size_t)(r0 + r) * N + n0]      = (bf16)c00[r];
      Yb[(size_t)(r0 + r) * N + n1]      = (bf16)c01[r];
      Yb[(size_t)(r0 + 16 + r) * N + n0] = (bf16)c10[r];
      Yb[(size_t)(r0 + 16 + r) * N + n1] = (bf16)c11[r];
    }
  }
}

// ---------------------------------------------------------------------------
// vT[b,h,d,t] = vh[b,t,h*64+d]  (coalesced reads; strided stores are
// fire-and-forget). Makes P@V B-fragments contiguous loads.
// ---------------------------------------------------------------------------
__global__ void transpose_v(const bf16* __restrict__ vh, bf16* __restrict__ vT) {
  int i = blockIdx.x * blockDim.x + threadIdx.x;
  if (i >= BB * SS * DD) return;
  int hd = i & (DD - 1);
  int t  = (i >> 10) & (SS - 1);
  int b  = i >> 21;
  int h  = hd >> 6;
  int d  = hd & (HD - 1);
  vT[(size_t)((b * HH + h) * HD + d) * SS + t] = vh[i];
}

// ---------------------------------------------------------------------------
// Attention core. One wave per (b, h, 16-query tile); the 4 waves of a block
// share (b,h) and adjacent query tiles, so K/V tiles are staged into LDS once
// per block with async-to-LDS copies.
// Pass 1: row max over ALL keys (reference maxes before causal mask).
// Pass 2: masked exp, P tile via LDS (C-layout -> A-layout), P@V WMMA.
// den = rowsum + 1 (quiet softmax).
// ---------------------------------------------------------------------------
__global__ void attn_core(const bf16* __restrict__ Qh, const bf16* __restrict__ Kh,
                          const bf16* __restrict__ VT, bf16* __restrict__ O) {
  __shared__ __attribute__((aligned(16))) bf16 kbuf[32 * 72];      // 32 keys x 64 d (pad 8)
  __shared__ __attribute__((aligned(16))) bf16 vbuf[64 * 40];      // 64 d x 32 keys (pad 8)
  __shared__ __attribute__((aligned(16))) bf16 pbuf[4 * 16 * 40];  // per-wave P tiles

  int tid  = threadIdx.x;
  int wl   = tid >> 5;
  int gw   = blockIdx.x * 4 + wl;
  int lane = tid & 31;
  int g = lane >> 4, l = lane & 15;
  int qt = gw & 127;
  int h  = (gw >> 7) & 15;
  int b  = gw >> 11;
  int qbase = qt * 16;
  const float scale = 0.125f;  // 1/sqrt(64)

  // Q fragments (d = 0..31, 32..63), reused for all score tiles.
  const bf16* qrow = Qh + (size_t)(b * SS + qbase + l) * DD + h * HD;
  v16bf qf0 = load_a_frag(qrow, g);
  v16bf qf1 = load_a_frag(qrow + 32, g);

  const bf16* kbase_p = Kh + (size_t)b * SS * DD + h * HD;
  const bf16* vbase_p = VT + (size_t)((b * HH + h) * HD) * SS;

  // ---- pass 1: unmasked row max over all S keys (block-shared K staging) ----
  float lm[8];
#pragma unroll
  for (int r = 0; r < 8; ++r) lm[r] = -3.0e38f;

  for (int jb = 0; jb < SS / 32; ++jb) {
    __syncthreads();  // previous tile's LDS readers done
#pragma unroll
    for (int c = tid; c < 256; c += 128) {      // 32 rows x 8 16B-chunks
      int row = c >> 3, ko = (c & 7) * 8;
      async_copy16((uint32_t)(uintptr_t)(kbuf + row * 72 + ko),
                   kbase_p + (size_t)(jb * 32 + row) * DD + ko);
    }
    wait_async0();
    __syncthreads();  // staging visible to all waves
#pragma unroll
    for (int half = 0; half < 2; ++half) {
      const bf16* kr = kbuf + (half * 16 + l) * 72;
      v8f c = {};
      c = wmma_bf16(qf0, load_b_frag(kr, g), c);
      c = wmma_bf16(qf1, load_b_frag(kr + 32, g), c);
#pragma unroll
      for (int r = 0; r < 8; ++r) lm[r] = fmaxf(lm[r], c[r] * scale);
    }
  }
#pragma unroll
  for (int m = 1; m < 16; m <<= 1)
#pragma unroll
    for (int r = 0; r < 8; ++r) lm[r] = fmaxf(lm[r], __shfl_xor(lm[r], m, 32));

  // ---- pass 2: masked exp + P@V (block-uniform trip count for barriers) ----
  float ls[8];
#pragma unroll
  for (int r = 0; r < 8; ++r) ls[r] = 0.f;
  v8f acc[4] = {v8f{}, v8f{}, v8f{}, v8f{}};
  bf16* pw = pbuf + wl * 640;                     // 16 rows x 40 bf16
  int nkb = ((qt | 3) * 16 + 16 + 31) >> 5;       // max causal range in block

  for (int jb = 0; jb < nkb; ++jb) {
    __syncthreads();
#pragma unroll
    for (int c = tid; c < 256; c += 128) {        // K tile: 32 x 64
      int row = c >> 3, ko = (c & 7) * 8;
      async_copy16((uint32_t)(uintptr_t)(kbuf + row * 72 + ko),
                   kbase_p + (size_t)(jb * 32 + row) * DD + ko);
    }
#pragma unroll
    for (int c = tid; c < 256; c += 128) {        // V tile: 64 d x 32 keys
      int d = c >> 2, ko = (c & 3) * 8;
      async_copy16((uint32_t)(uintptr_t)(vbuf + d * 40 + ko),
                   vbase_p + (size_t)d * SS + jb * 32 + ko);
    }
    wait_async0();
    __syncthreads();

#pragma unroll
    for (int half = 0; half < 2; ++half) {
      int kt16 = jb * 2 + half;
      const bf16* kr = kbuf + (half * 16 + l) * 72;
      v8f c = {};
      c = wmma_bf16(qf0, load_b_frag(kr, g), c);
      c = wmma_bf16(qf1, load_b_frag(kr + 32, g), c);
#pragma unroll
      for (int r = 0; r < 8; ++r) {
        int rowg = qbase + r + 8 * g;
        int colg = kt16 * 16 + l;
        float p = 0.f;
        if (colg <= rowg) {  // causal mask (also zeroes block-padding tiles)
          p = __expf(c[r] * scale - lm[r]);
          ls[r] += p;
        }
        pw[(r + 8 * g) * 40 + half * 16 + l] = (bf16)p;
      }
    }
    asm volatile("s_wait_dscnt 0" ::: "memory");  // same-wave LDS turn-around
    v16bf ap = load_a_frag(pw + l * 40, g);       // P as A-fragment (16x32)
#pragma unroll
    for (int dc = 0; dc < 4; ++dc) {
      const bf16* vr = vbuf + (dc * 16 + l) * 40;
      acc[dc] = wmma_bf16(ap, load_b_frag(vr, g), acc[dc]);
    }
  }

  // row sums -> quiet-softmax denominator
#pragma unroll
  for (int m = 1; m < 16; m <<= 1)
#pragma unroll
    for (int r = 0; r < 8; ++r) ls[r] += __shfl_xor(ls[r], m, 32);
  float inv[8];
#pragma unroll
  for (int r = 0; r < 8; ++r) inv[r] = 1.0f / (ls[r] + 1.0f);

#pragma unroll
  for (int dc = 0; dc < 4; ++dc)
#pragma unroll
    for (int r = 0; r < 8; ++r) {
      int row = qbase + r + 8 * g;
      int col = h * HD + dc * 16 + l;
      O[(size_t)(b * SS + row) * DD + col] = (bf16)(acc[dc][r] * inv[r]);
    }
}

// ---------------------------------------------------------------------------
// launch
// ---------------------------------------------------------------------------
extern "C" void kernel_launch(void* const* d_in, const int* in_sizes, int n_in,
                              void* d_out, int out_size, void* d_ws, size_t ws_size,
                              hipStream_t stream) {
  const float* q  = (const float*)d_in[0];
  const float* k  = (const float*)d_in[1];
  const float* v  = (const float*)d_in[2];
  const float* Wq = (const float*)d_in[3];
  const float* Wk = (const float*)d_in[4];
  const float* Wv = (const float*)d_in[5];
  const float* Wo = (const float*)d_in[6];

  const size_t NQ = (size_t)BB * SS * DD;  // 4M elems
  const size_t NW = (size_t)DD * DD;       // 1M elems

  bf16* ws  = (bf16*)d_ws;
  bf16* qb  = ws;
  bf16* kb  = qb + NQ;
  bf16* vb  = kb + NQ;
  bf16* wqb = vb + NQ;
  bf16* wkb = wqb + NW;
  bf16* wvb = wkb + NW;
  bf16* wob = wvb + NW;
  bf16* qh  = wob + NW;
  bf16* kh  = qh + NQ;
  bf16* vh  = kh + NQ;
  bf16* vT  = vh + NQ;
  bf16* ao  = vT + NQ;

  int nqb = (int)((NQ + 255) / 256);
  int nwb = (int)((NW + 255) / 256);
  cvt_f32_bf16<<<nqb, 256, 0, stream>>>(q, qb, (int)NQ);
  cvt_f32_bf16<<<nqb, 256, 0, stream>>>(k, kb, (int)NQ);
  cvt_f32_bf16<<<nqb, 256, 0, stream>>>(v, vb, (int)NQ);
  cvt_f32_bf16<<<nwb, 256, 0, stream>>>(Wq, wqb, (int)NW);
  cvt_f32_bf16<<<nwb, 256, 0, stream>>>(Wk, wkb, (int)NW);
  cvt_f32_bf16<<<nwb, 256, 0, stream>>>(Wv, wvb, (int)NW);
  cvt_f32_bf16<<<nwb, 256, 0, stream>>>(Wo, wob, (int)NW);

  // projections: 4096x1024 @ 1024x1024^T -> (128 x 32) 32x32 tiles, 8 waves/blk
  const int M = BB * SS, N = DD, K = DD;
  const int gemm_blocks = (M / 32) * (N / 32) / 8;  // 512
  gemm_xwt<<<gemm_blocks, 256, 0, stream>>>(qb, wqb, qh, nullptr, M, N, K);
  gemm_xwt<<<gemm_blocks, 256, 0, stream>>>(kb, wkb, kh, nullptr, M, N, K);
  gemm_xwt<<<gemm_blocks, 256, 0, stream>>>(vb, wvb, vh, nullptr, M, N, K);

  transpose_v<<<nqb, 256, 0, stream>>>(vh, vT);

  // 2*16*128 = 4096 waves, 4 waves/block sharing (b,h)
  attn_core<<<1024, 128, 0, stream>>>(qh, kh, vT, ao);

  // final projection -> f32 output
  gemm_xwt<<<gemm_blocks, 256, 0, stream>>>(ao, wob, nullptr, (float*)d_out, M, N, K);
}
